// MultiHeadAttention_21466246546080
// MI455X (gfx1250) — matrix operand order repaired and checked by reading, not compiled, not run on hardware
//
#include <hip/hip_runtime.h>

// ---------------------------------------------------------------------------
// MHA forward for MI455X (gfx1250, wave32, WMMA 16x16x32 bf16, fp32 accum)
//   B=4, T=2048, D_MODEL=1024, H=16, d_k=64
// Pipeline:
//   1) convert x, W_q, W_k, W_v, W_o  fp32 -> bf16 (workspace)
//   2) Q = (x Wq^T) * 0.125, K = x Wk^T   (WMMA GEMM, bf16 out, [B*T, D])
//      Vt = (x Wv^T) stored [B,H,dk,T]    (WMMA GEMM, transposed bf16 store)
//   3) causal flash attention (WMMA QK^T, online softmax, WMMA P.V)
//   4) out = attn Wo^T                    (WMMA GEMM, fp32 out -> d_out)
// Data movement: A-tile via GLOBAL_LOAD_ASYNC_TO_LDS_B128 (ASYNCcnt),
//                B-tile via TDM tensor_load_to_lds (TENSORcnt, LDS padding),
//                with synchronous fallbacks if builtins are unavailable.
// ---------------------------------------------------------------------------

typedef __attribute__((ext_vector_type(16))) __bf16 v16bf;
typedef __attribute__((ext_vector_type(8)))  __bf16 v8bf;
typedef __attribute__((ext_vector_type(4)))  __bf16 v4bf;
typedef __attribute__((ext_vector_type(8)))  float  v8f;
typedef __attribute__((ext_vector_type(4)))  unsigned int v4u;
typedef __attribute__((ext_vector_type(8)))  int v8i;
typedef __attribute__((ext_vector_type(4)))  int v4i;

#define T_SEQ   2048
#define N_HEADS 16
#define D_K     64

#if __has_builtin(__builtin_amdgcn_global_load_async_to_lds_b128)
#define HAS_ASYNC_LDS 1
#else
#define HAS_ASYNC_LDS 0
#endif
#if __has_builtin(__builtin_amdgcn_tensor_load_to_lds) && \
    __has_builtin(__builtin_amdgcn_s_wait_tensorcnt)
#define HAS_TDM 1
#else
#define HAS_TDM 0
#endif

// Assemble a 16-element bf16 fragment from two 16-byte chunks.
__device__ __forceinline__ v16bf frag16(const __bf16* p0, const __bf16* p1) {
  union { v16bf v; v8bf h[2]; } u;
  u.h[0] = *(const v8bf*)p0;
  u.h[1] = *(const v8bf*)p1;
  return u.v;
}

__device__ __forceinline__ void wait_lds() {
#if __has_builtin(__builtin_amdgcn_s_wait_dscnt)
  __builtin_amdgcn_s_wait_dscnt(0);
#else
  asm volatile("s_wait_dscnt 0" ::: "memory");
#endif
}

__device__ __forceinline__ void wait_async() {
#if __has_builtin(__builtin_amdgcn_s_wait_asynccnt)
  __builtin_amdgcn_s_wait_asynccnt(0);
#else
  asm volatile("s_wait_asynccnt 0" ::: "memory");
#endif
}

// 16-lane xor-butterfly reductions. v_permlane16_b32 keeps these in the VALU
// pipe (selectors encode lane^m within each 16-lane row); fallback ds_bpermute.
#if __has_builtin(__builtin_amdgcn_permlane16)
#define ROW16_STEP(v, OPC, s0, s1)                                            \
  {                                                                           \
    unsigned _u = __builtin_amdgcn_permlane16(                                \
        __float_as_uint(v), __float_as_uint(v), s0, s1, false, false);        \
    v = OPC(v, __uint_as_float(_u));                                          \
  }
__device__ __forceinline__ float red16_max(float v) {
  ROW16_STEP(v, fmaxf, 0x67452301u, 0xEFCDAB89u)   // xor 1
  ROW16_STEP(v, fmaxf, 0x45670123u, 0xCDEF89ABu)   // xor 3
  ROW16_STEP(v, fmaxf, 0x32107654u, 0xBA98FEDCu)   // xor 4
  ROW16_STEP(v, fmaxf, 0xFEDCBA98u, 0x76543210u)   // xor 8
  return v;
}
__device__ __forceinline__ float addf(float a, float b) { return a + b; }
__device__ __forceinline__ float red16_sum(float v) {
  ROW16_STEP(v, addf, 0x67452301u, 0xEFCDAB89u)
  ROW16_STEP(v, addf, 0x45670123u, 0xCDEF89ABu)
  ROW16_STEP(v, addf, 0x32107654u, 0xBA98FEDCu)
  ROW16_STEP(v, addf, 0xFEDCBA98u, 0x76543210u)
  return v;
}
#else
__device__ __forceinline__ float red16_max(float v) {
  #pragma unroll
  for (int m = 1; m < 16; m <<= 1) v = fmaxf(v, __shfl_xor(v, m, 32));
  return v;
}
__device__ __forceinline__ float red16_sum(float v) {
  #pragma unroll
  for (int m = 1; m < 16; m <<= 1) v += __shfl_xor(v, m, 32);
  return v;
}
#endif

#if HAS_ASYNC_LDS
// Builtin signature (probed via diagnostics): (v4i AS1* src, v4i AS3* lds,
// imm offset, imm cpol).
__device__ __forceinline__ void async_copy16(const __bf16* g, __bf16* l) {
  __builtin_amdgcn_global_load_async_to_lds_b128(
      (__attribute__((address_space(1))) v4i*)(v4i*)(void*)g,
      (__attribute__((address_space(3))) v4i*)(v4i*)(void*)l, 0, 0);
}
#endif

#if HAS_TDM
__device__ __forceinline__ unsigned lds_addr_of(const void* p) {
  return (unsigned)(unsigned long long)
      (__attribute__((address_space(3))) const void*)p;
}
// 2-D TDM load: tile_rows x tile_cols bf16, row pitch = tensor_cols elements.
// Pads the LDS destination: every 64 B (16 dwords) insert 16 B (4 dwords),
// producing a 40-element LDS row stride for a 32-element tile row.
__device__ __forceinline__ void tdm_load_2d(const void* gsrc, unsigned ldsaddr,
                                            unsigned tensor_cols,
                                            unsigned tensor_rows,
                                            unsigned tile_cols,
                                            unsigned tile_rows) {
  unsigned long long ga = (unsigned long long)gsrc;
  v4u g0 = {1u,  // count=1, user descriptor, no gather
            ldsaddr, (unsigned)ga,
            (unsigned)((ga >> 32) & 0x01ffffffull) | (2u << 30)};  // type=2
  v8i g1;
  g1[0] = (int)((1u << 16) |        // data_size = 2 bytes
                (1u << 20) |        // pad_enable
                (3u << 22) |        // pad_interval: 16 dwords
                (3u << 25));        // pad_amount:   4 dwords
  g1[1] = (int)((tensor_cols & 0xffffu) << 16);
  g1[2] = (int)(((tensor_cols >> 16) & 0xffffu) |
                ((tensor_rows & 0xffffu) << 16));
  g1[3] = (int)(((tensor_rows >> 16) & 0xffffu) |
                ((tile_cols & 0xffffu) << 16));
  g1[4] = (int)(tile_rows & 0xffffu);  // tile_dim1; tile_dim2 = 0
  g1[5] = (int)tensor_cols;            // dim0 stride lo32
  g1[6] = 0;                           // dim0 stride hi16 | dim1 stride lo16
  g1[7] = 0;
  v4i z4 = {0, 0, 0, 0};
#if defined(__clang_major__) && (__clang_major__ >= 23)
  v8i z8 = {0, 0, 0, 0, 0, 0, 0, 0};
  __builtin_amdgcn_tensor_load_to_lds(g0, g1, z4, z4, z8, 0);
#else
  __builtin_amdgcn_tensor_load_to_lds(g0, g1, z4, z4, 0);
#endif
}
#endif

// ---------------------------------------------------------------------------
// fp32 -> bf16 convert (n multiple of 4, grid sized exactly)
// ---------------------------------------------------------------------------
__global__ __launch_bounds__(256) void cvt_f32_bf16(
    const float* __restrict__ s, __bf16* __restrict__ d, int n) {
  int i = (blockIdx.x * 256 + threadIdx.x) * 4;
  if (i + 3 < n) {
    float4 f = *(const float4*)(s + i);
    v4bf o = {(__bf16)f.x, (__bf16)f.y, (__bf16)f.z, (__bf16)f.w};
    *(v4bf*)(d + i) = o;
  }
}

// ---------------------------------------------------------------------------
// WMMA GEMM: C[M,N] = A[M,K] * B[N,K]^T   (both row-major over K)
// Block tile 128x128, BK=32, 8 waves (4M x 2N), each wave 32x64 = 8 WMMA accs.
// Double-buffered LDS (row stride 40 bf16). A-tile staged with async-to-LDS,
// B-tile staged with one TDM descriptor from wave 0.
// ---------------------------------------------------------------------------
enum { OUT_F32 = 0, OUT_BF16 = 1, OUT_BF16_VT = 2, OUT_BF16_QS = 3 };

template <int MODE>
__global__ __launch_bounds__(256) void gemm_bf16_nt(
    const __bf16* __restrict__ A,  // [M,K]
    const __bf16* __restrict__ B,  // [N,K]
    void* __restrict__ Cout,
    int M, int N, int K) {
  __shared__ __bf16 sA[2][128 * 40];
  __shared__ __bf16 sB[2][128 * 40];

  const int m0   = blockIdx.y * 128;
  const int n0   = blockIdx.x * 128;
  const int tid  = threadIdx.x;
  const int lane = tid & 31;
  const int wave = tid >> 5;
  const int wm   = (wave & 3) * 32;   // wave M offset inside tile
  const int wn   = (wave >> 2) * 64;  // wave N offset inside tile
  const int r    = lane & 15;         // row (A) / column (B) owner
  const int kh   = lane >> 4;         // K half select

  v8f acc[2][4] = {};
  const int ntile = K >> 5;

  // ---- stage k-tile 0 ----
  {
    #pragma unroll
    for (int c = 0; c < 2; ++c) {
      int chunk = tid + c * 256;  // 512 chunks of 8 bf16 for the A tile
      int row = chunk >> 2;
      int col = (chunk & 3) << 3;
      const __bf16* g = &A[(size_t)(m0 + row) * K + col];
      __bf16* l = &sA[0][row * 40 + col];
#if HAS_ASYNC_LDS
      async_copy16(g, l);
#else
      *(v8bf*)l = *(const v8bf*)g;
#endif
    }
#if HAS_TDM
    if (wave == 0)
      tdm_load_2d(&B[(size_t)n0 * K], lds_addr_of(&sB[0][0]),
                  (unsigned)K, (unsigned)(N - n0), 32u, 128u);
#else
    #pragma unroll
    for (int c = 0; c < 2; ++c) {
      int chunk = tid + c * 256;
      int row = chunk >> 2;
      int col = (chunk & 3) << 3;
      const __bf16* g = &B[(size_t)(n0 + row) * K + col];
      __bf16* l = &sB[0][row * 40 + col];
#if HAS_ASYNC_LDS
      async_copy16(g, l);
#else
      *(v8bf*)l = *(const v8bf*)g;
#endif
    }
#endif
#if HAS_ASYNC_LDS
    wait_async();
#endif
#if HAS_TDM
    __builtin_amdgcn_s_wait_tensorcnt(0);
#endif
    __syncthreads();
  }

  for (int kt = 0; kt < ntile; ++kt) {
    const int cur = kt & 1;
    if (kt + 1 < ntile) {
      const int nxt = cur ^ 1;
      const int k0  = (kt + 1) << 5;
      #pragma unroll
      for (int c = 0; c < 2; ++c) {
        int chunk = tid + c * 256;
        int row = chunk >> 2;
        int col = (chunk & 3) << 3;
        const __bf16* g = &A[(size_t)(m0 + row) * K + k0 + col];
        __bf16* l = &sA[nxt][row * 40 + col];
#if HAS_ASYNC_LDS
        async_copy16(g, l);
#else
        *(v8bf*)l = *(const v8bf*)g;
#endif
      }
#if HAS_TDM
      if (wave == 0)
        tdm_load_2d(&B[(size_t)n0 * K + k0], lds_addr_of(&sB[nxt][0]),
                    (unsigned)K, (unsigned)(N - n0), 32u, 128u);
#else
      #pragma unroll
      for (int c = 0; c < 2; ++c) {
        int chunk = tid + c * 256;
        int row = chunk >> 2;
        int col = (chunk & 3) << 3;
        const __bf16* g = &B[(size_t)(n0 + row) * K + k0 + col];
        __bf16* l = &sB[nxt][row * 40 + col];
#if HAS_ASYNC_LDS
        async_copy16(g, l);
#else
        *(v8bf*)l = *(const v8bf*)g;
#endif
      }
#endif
    }

    const __bf16* sa = sA[cur];
    const __bf16* sb = sB[cur];

    // A fragments: lane pair (r, r+16) shares row; kh selects K sub-ranges.
    v16bf af[2];
    #pragma unroll
    for (int mi = 0; mi < 2; ++mi) {
      const __bf16* p = sa + (wm + mi * 16 + r) * 40 + 16 * kh;
      af[mi] = frag16(p, p + 8);
    }
    // B fragments: lane owns column r, elements run over K = 16*kh + e.
    v16bf bfr[4];
    #pragma unroll
    for (int ni = 0; ni < 4; ++ni) {
      const __bf16* p = sb + (wn + ni * 16 + r) * 40 + 16 * kh;
      bfr[ni] = frag16(p, p + 8);
    }

    #pragma unroll
    for (int mi = 0; mi < 2; ++mi)
      #pragma unroll
      for (int ni = 0; ni < 4; ++ni)
        acc[mi][ni] = __builtin_amdgcn_wmma_f32_16x16x32_bf16(
            false, af[mi], false, bfr[ni], (short)0, acc[mi][ni], false, false);

#if HAS_ASYNC_LDS
    wait_async();
#endif
#if HAS_TDM
    __builtin_amdgcn_s_wait_tensorcnt(0);
#endif
    __syncthreads();
  }

  // Store: C layout -> acc element i is row (i + 8*kh), lane r is the column.
  #pragma unroll
  for (int mi = 0; mi < 2; ++mi) {
    #pragma unroll
    for (int ni = 0; ni < 4; ++ni) {
      #pragma unroll
      for (int i = 0; i < 8; ++i) {
        const int   mg = m0 + wm + mi * 16 + i + 8 * kh;
        const int   ng = n0 + wn + ni * 16 + r;
        const float v  = acc[mi][ni][i];
        if (MODE == OUT_F32) {
          ((float*)Cout)[(size_t)mg * N + ng] = v;
        } else if (MODE == OUT_BF16) {
          ((__bf16*)Cout)[(size_t)mg * N + ng] = (__bf16)v;
        } else if (MODE == OUT_BF16_QS) {  // fold 1/sqrt(d_k) into Q
          ((__bf16*)Cout)[(size_t)mg * N + ng] = (__bf16)(v * 0.125f);
        } else {  // OUT_BF16_VT: [B*T, D] -> [B, H, d_k, T]
          const int b = mg / T_SEQ, t = mg % T_SEQ;
          const int h = ng >> 6, d = ng & 63;
          ((__bf16*)Cout)[(((size_t)b * N_HEADS + h) * D_K + d) * T_SEQ + t] =
              (__bf16)v;
        }
      }
    }
  }
}

// ---------------------------------------------------------------------------
// Causal flash attention. One wave owns a 16-query tile; 4 waves per block.
// Q (pre-scaled), K: [B*T, 1024] bf16 (head h at column h*64).
// Vt: [B,H,d_k,T] bf16.
// ---------------------------------------------------------------------------
__global__ __launch_bounds__(128) void flash_attn_fwd(
    const __bf16* __restrict__ Q, const __bf16* __restrict__ K,
    const __bf16* __restrict__ Vt, __bf16* __restrict__ O) {
  __shared__ __bf16 sP[4][16 * 40];  // per-wave P staging (C-layout->A-layout)

  const int lane = threadIdx.x & 31;
  const int wave = threadIdx.x >> 5;
  const int r    = lane & 15;
  const int kh   = lane >> 4;
  const int q0   = blockIdx.x * 64 + wave * 16;
  const int b    = blockIdx.y >> 4;
  const int h    = blockIdx.y & 15;

  // Q A-fragments for d = 0..31 and 32..63
  const __bf16* qrow = Q + ((size_t)(b * T_SEQ + q0 + r)) * 1024 + h * 64;
  v16bf aq[2];
  #pragma unroll
  for (int j = 0; j < 2; ++j) {
    const __bf16* p = qrow + 32 * j + 16 * kh;
    aq[j] = frag16(p, p + 8);
  }

  v8f   o[4] = {};  // 16 x 64 output accumulator
  float mrow[8], lrow[8];
  #pragma unroll
  for (int i = 0; i < 8; ++i) { mrow[i] = -3.0e38f; lrow[i] = 0.0f; }

  __bf16* sp = sP[wave];

  for (int kt0 = 0; kt0 <= q0 + 15; kt0 += 32) {
    // ---- S = Q K^T for 32 keys (two 16-key column blocks) ----
    v8f s0 = {}, s1 = {};
    #pragma unroll
    for (int j = 0; j < 2; ++j) {
      const __bf16* kp0 =
          K + ((size_t)(b * T_SEQ + kt0 + r)) * 1024 + h * 64 + 32 * j + 16 * kh;
      v16bf kb0 = frag16(kp0, kp0 + 8);
      s0 = __builtin_amdgcn_wmma_f32_16x16x32_bf16(false, aq[j], false, kb0,
                                                   (short)0, s0, false, false);
      const __bf16* kp1 = K + ((size_t)(b * T_SEQ + kt0 + 16 + r)) * 1024 +
                          h * 64 + 32 * j + 16 * kh;
      v16bf kb1 = frag16(kp1, kp1 + 8);
      s1 = __builtin_amdgcn_wmma_f32_16x16x32_bf16(false, aq[j], false, kb1,
                                                   (short)0, s1, false, false);
    }

    // ---- causal mask, online softmax; stage P into LDS ----
    #pragma unroll
    for (int i = 0; i < 8; ++i) {
      const int row = q0 + i + 8 * kh;
      float a0 = s0[i]; if (kt0 + r > row)      a0 = -3.0e38f;
      float a1 = s1[i]; if (kt0 + 16 + r > row) a1 = -3.0e38f;
      const float rm    = red16_max(fmaxf(a0, a1));
      const float mn    = fmaxf(mrow[i], rm);
      const float alpha = __expf(mrow[i] - mn);
      const float p0    = __expf(a0 - mn);
      const float p1    = __expf(a1 - mn);
      const float rs    = red16_sum(p0 + p1);
      lrow[i] = lrow[i] * alpha + rs;
      mrow[i] = mn;
      #pragma unroll
      for (int j = 0; j < 4; ++j) o[j][i] *= alpha;
      sp[(i + 8 * kh) * 40 + r]      = (__bf16)p0;
      sp[(i + 8 * kh) * 40 + 16 + r] = (__bf16)p1;
    }
    wait_lds();

    // ---- O += P V (P re-read in A-layout; Vt gives contiguous B-frags) ----
    const __bf16* pp = sp + r * 40 + 16 * kh;
    v16bf ap = frag16(pp, pp + 8);
    #pragma unroll
    for (int j = 0; j < 4; ++j) {
      const __bf16* vp = Vt +
          ((size_t)(b * N_HEADS + h) * D_K + 16 * j + r) * T_SEQ + kt0 + 16 * kh;
      v16bf vb = frag16(vp, vp + 8);
      o[j] = __builtin_amdgcn_wmma_f32_16x16x32_bf16(false, ap, false, vb,
                                                     (short)0, o[j], false, false);
    }
  }

  // ---- normalize and store attn output [B*T, 1024] bf16 ----
  #pragma unroll
  for (int i = 0; i < 8; ++i) {
    const float inv = 1.0f / lrow[i];
    const int   row = q0 + i + 8 * kh;
    __bf16* orow = O + ((size_t)(b * T_SEQ + row)) * 1024 + h * 64;
    #pragma unroll
    for (int j = 0; j < 4; ++j) orow[16 * j + r] = (__bf16)(o[j][i] * inv);
  }
}

// ---------------------------------------------------------------------------
// Host launcher
// ---------------------------------------------------------------------------
extern "C" void kernel_launch(void* const* d_in, const int* in_sizes, int n_in,
                              void* d_out, int out_size, void* d_ws,
                              size_t ws_size, hipStream_t stream) {
  (void)in_sizes; (void)n_in; (void)out_size; (void)ws_size;
  const float* x  = (const float*)d_in[0];
  const float* Wq = (const float*)d_in[1];
  const float* Wk = (const float*)d_in[2];
  const float* Wv = (const float*)d_in[3];
  const float* Wo = (const float*)d_in[4];
  float*       out = (float*)d_out;

  const int B = 4, T = T_SEQ, D = 1024;
  const int MT = B * T;  // 8192

  char* ws = (char*)d_ws;
  __bf16* xb  = (__bf16*)(ws);                          // 16 MB  x bf16
  __bf16* wqb = (__bf16*)(ws + (size_t)(16) * 1048576); //  2 MB
  __bf16* wkb = (__bf16*)(ws + (size_t)(18) * 1048576);
  __bf16* wvb = (__bf16*)(ws + (size_t)(20) * 1048576);
  __bf16* wob = (__bf16*)(ws + (size_t)(22) * 1048576);
  __bf16* Qb  = (__bf16*)(ws + (size_t)(24) * 1048576); // 16 MB [B*T, D]
  __bf16* Kb  = (__bf16*)(ws + (size_t)(40) * 1048576); // 16 MB [B*T, D]
  __bf16* Vtb = (__bf16*)(ws + (size_t)(56) * 1048576); // 16 MB [B,H,dk,T]
  __bf16* Ab  = (__bf16*)(ws + (size_t)(72) * 1048576); // 16 MB attn out

  cvt_f32_bf16<<<(MT * D) / 1024, 256, 0, stream>>>(x, xb, MT * D);
  cvt_f32_bf16<<<(D * D) / 1024, 256, 0, stream>>>(Wq, wqb, D * D);
  cvt_f32_bf16<<<(D * D) / 1024, 256, 0, stream>>>(Wk, wkb, D * D);
  cvt_f32_bf16<<<(D * D) / 1024, 256, 0, stream>>>(Wv, wvb, D * D);
  cvt_f32_bf16<<<(D * D) / 1024, 256, 0, stream>>>(Wo, wob, D * D);

  dim3 gg(D / 128, MT / 128);  // (8, 64)
  gemm_bf16_nt<OUT_BF16_QS><<<gg, 256, 0, stream>>>(xb, wqb, Qb, MT, D, D);
  gemm_bf16_nt<OUT_BF16><<<gg, 256, 0, stream>>>(xb, wkb, Kb, MT, D, D);
  gemm_bf16_nt<OUT_BF16_VT><<<gg, 256, 0, stream>>>(xb, wvb, Vtb, MT, D, D);

  flash_attn_fwd<<<dim3(T / 64, B * N_HEADS), 128, 0, stream>>>(Qb, Kb, Vtb, Ab);

  gemm_bf16_nt<OUT_F32><<<gg, 256, 0, stream>>>(Ab, wob, out, MT, D, D);
}
